// AttentionObstacleEncoder_46377056862217
// MI455X (gfx1250) — compile-verified, matrix operand order checked by reading, not run on hardware
//
#include <hip/hip_runtime.h>

// ---------------- constants ----------------
#define DD   256   // latent dim
#define NN   64    // max obstacles
#define HH   4     // heads
#define HDIM 64    // head dim
#define SD   264   // half stride for row-major [64][SD] bufs (528B, 16B-aligned rows)
#define SVD  72    // half stride for transposed-v / attn scratch (144B rows)
#define SWD  40    // half stride for staged weight chunk [256][SWD] (80B rows)
#define SAW  (16*SVD)  // per-wave attn scratch (halves)
#define NEGBIG (-3.402823466e38f)

typedef __attribute__((ext_vector_type(16))) _Float16 v16h;
typedef __attribute__((ext_vector_type(8)))  _Float16 v8h;
typedef __attribute__((ext_vector_type(8)))  float    v8f;

union Frag { v16h v; v8h h[2]; };
union Acc  { v8f  v; float f[8]; };

// A/B fragment load: lane-local base already includes row*stride + kchunk + (lane/16)*8.
// Pattern per ISA: halves [0..7] and [16..23] relative to that base.
__device__ __forceinline__ v16h load_frag(const _Float16* p) {
  Frag f;
  f.h[0] = *(const v8h*)(p);
  f.h[1] = *(const v8h*)(p + 16);
  return f.v;
}

// One [64,256] x [256,256] GEMM, 8 waves.
// Wave w covers ALL 4 M-tiles x N-tiles {2w, 2w+1}: each B fragment feeds
// 4 back-to-back WMMAs; columns 32w..32w+31 are wave-private.
// MODE 0: Dst[row][col] += acc          (row-major f16, dest prefilled e.g. pos+b2)
// MODE 1: Dst[row][col]  = acc + bias   (row-major f16)
// MODE 2: Dst[col][row]  = acc + bias   (transposed f16)
// MODE 3: masked mean/max pooling into sPool (no atomics: columns wave-private)
template<int MODE>
__device__ __forceinline__ void gemm_step(
    const _Float16* __restrict__ Asrc, const float* __restrict__ Wg,
    const float* __restrict__ nextWg,
    _Float16* Dst, _Float16* sWt, const float* __restrict__ bias,
    float* sPool, int len)
{
  const int tid   = threadIdx.x;
  const int lane  = tid & 31;
  const int wv_   = tid >> 5;
  const int lmod  = lane & 15;
  const int lhalf = lane >> 4;
  const int koff  = lhalf * 8;
  const int NtB   = wv_ * 2;

  Acc acc[4][2];
  #pragma unroll
  for (int mt = 0; mt < 4; ++mt)
    #pragma unroll
    for (int n = 0; n < 2; ++n)
      #pragma unroll
      for (int j = 0; j < 8; ++j) acc[mt][n].f[j] = 0.f;

  for (int kk = 0; kk < 8; ++kk) {
    __syncthreads();
    // stage W[kk*32 .. +31][0..255] -> sWt[n][r] (f16, transposed);
    // prefetch the next chunk (or the next GEMM's first chunk).
    const float* pf = (kk < 7) ? (Wg + (kk + 1) * 32 * DD) : nextWg;
    #pragma unroll 8
    for (int r = 0; r < 32; ++r) {
      sWt[tid * SWD + r] = (_Float16)Wg[(kk * 32 + r) * DD + tid];
      __builtin_prefetch(pf + r * DD + tid, 0, 3);
    }
    __syncthreads();

    v16h fa[4];
    #pragma unroll
    for (int mt = 0; mt < 4; ++mt)
      fa[mt] = load_frag(Asrc + (mt * 16 + lmod) * SD + kk * 32 + koff);
    #pragma unroll
    for (int nt = 0; nt < 2; ++nt) {
      v16h fb = load_frag(sWt + ((NtB + nt) * 16 + lmod) * SWD + koff);
      #pragma unroll
      for (int mt = 0; mt < 4; ++mt)
        acc[mt][nt].v = __builtin_amdgcn_wmma_f32_16x16x32_f16(
            false, fa[mt], false, fb, (short)0, acc[mt][nt].v, false, false);
    }
  }

  if constexpr (MODE == 0) {
    #pragma unroll
    for (int mt = 0; mt < 4; ++mt)
      #pragma unroll
      for (int nt = 0; nt < 2; ++nt) {
        const int col = (NtB + nt) * 16 + lmod;
        #pragma unroll
        for (int r = 0; r < 8; ++r) {
          const int row = mt * 16 + lhalf * 8 + r;
          _Float16* p = Dst + row * SD + col;
          *p = (_Float16)((float)(*p) + acc[mt][nt].f[r]);
        }
      }
  } else if constexpr (MODE == 1 || MODE == 2) {
    #pragma unroll
    for (int mt = 0; mt < 4; ++mt)
      #pragma unroll
      for (int nt = 0; nt < 2; ++nt) {
        const int col = (NtB + nt) * 16 + lmod;
        const float bb = bias[col];
        #pragma unroll
        for (int r = 0; r < 8; ++r) {
          const int row = mt * 16 + lhalf * 8 + r;
          _Float16* p = (MODE == 1) ? (Dst + row * SD + col)
                                    : (Dst + col * SVD + row);
          *p = (_Float16)(acc[mt][nt].f[r] + bb);
        }
      }
  } else {
    // full 64-row masked mean/max per wave-private column; no atomics
    #pragma unroll
    for (int nt = 0; nt < 2; ++nt) {
      const int col = (NtB + nt) * 16 + lmod;
      const float bb = bias[col];
      float s = 0.f, mx = NEGBIG;
      #pragma unroll
      for (int mt = 0; mt < 4; ++mt)
        #pragma unroll
        for (int r = 0; r < 8; ++r) {
          const int row = mt * 16 + lhalf * 8 + r;
          const float v = acc[mt][nt].f[r] + bb;
          const bool ok = row < len;
          s  += ok ? v : 0.f;
          mx  = ok ? fmaxf(mx, v) : mx;
        }
      s  += __shfl_xor(s, 16);
      mx  = fmaxf(mx, __shfl_xor(mx, 16));
      if (lane < 16) {
        sPool[col]      = s;
        sPool[DD + col] = mx;
      }
    }
  }
}

__global__ void __launch_bounds__(256, 1) __attribute__((amdgpu_waves_per_eu(2)))
obstacle_encoder_kernel(
    const float* __restrict__ obstacles, const int* __restrict__ lengths,
    const float* __restrict__ w1,  const float* __restrict__ b1,
    const float* __restrict__ w2,  const float* __restrict__ b2,
    const float* __restrict__ pos,
    const float* __restrict__ wq,  const float* __restrict__ bq,
    const float* __restrict__ wk,  const float* __restrict__ bk,
    const float* __restrict__ wv,  const float* __restrict__ bv,
    const float* __restrict__ wo,  const float* __restrict__ bo,
    const float* __restrict__ g1w, const float* __restrict__ g1b,
    const float* __restrict__ g2w, const float* __restrict__ g2b,
    const float* __restrict__ ln_g, const float* __restrict__ ln_b,
    float* __restrict__ out)
{
  __shared__ _Float16 sA[NN * SD];            // h1 -> q
  __shared__ _Float16 sB[NN * SD];            // pos+b2 -> h -> ctx
  __shared__ _Float16 sC[NN * SD];            // k
  __shared__ _Float16 sVt[DD * SVD];          // v transposed [dim][key]
  __shared__ __align__(16) char sU[DD * SWD * 2];  // union: sWt / sX / sAttn / sG
  __shared__ float sPool[2 * DD];
  __shared__ float sRed[16];

  _Float16* sWt   = (_Float16*)sU;
  float*    sX    = (float*)sU;
  _Float16* sAttn = (_Float16*)sU;
  float*    sG    = (float*)sU;

  const int tid   = threadIdx.x;
  const int b     = blockIdx.x;
  const int lane  = tid & 31;
  const int wv_   = tid >> 5;
  const int lmod  = lane & 15;
  const int lhalf = lane >> 4;
  const int koff  = lhalf * 8;
  const int c     = tid;          // this thread's column for vector phases
  const int len   = lengths[b];

  // ---------- phase 0: obstacle MLP layer 1 (K=4, VALU) + prefill h dest ----------
  sX[tid] = obstacles[(size_t)b * NN * 4 + tid];   // [64][4] staged
  __builtin_prefetch(w2 + tid, 0, 3);              // warm first chunk of w2
  __builtin_prefetch(w2 + 8 * DD + tid, 0, 3);
  __builtin_prefetch(w2 + 16 * DD + tid, 0, 3);
  __builtin_prefetch(w2 + 24 * DD + tid, 0, 3);
  __syncthreads();
  {
    const float w1c0 = w1[0 * DD + c], w1c1 = w1[1 * DD + c];
    const float w1c2 = w1[2 * DD + c], w1c3 = w1[3 * DD + c];
    const float b1c = b1[c], b2c = b2[c];
    #pragma unroll 4
    for (int m = 0; m < NN; ++m) {
      float a = fmaf(sX[m * 4 + 0], w1c0, b1c);
      a = fmaf(sX[m * 4 + 1], w1c1, a);
      a = fmaf(sX[m * 4 + 2], w1c2, a);
      a = fmaf(sX[m * 4 + 3], w1c3, a);
      sA[m * SD + c] = (_Float16)fmaxf(a, 0.f);                 // h1 (relu)
      sB[m * SD + c] = (_Float16)(pos[m * DD + c] + b2c);       // prefill: pos + b2
    }
  }
  // (gemm's first barrier protects sX -> sWt reuse)

  // ---------- h = relu(x@w1+b1)@w2 + b2 + pos ----------
  gemm_step<0>(sA, w2, wq, sB, sWt, nullptr, nullptr, len);

  // ---------- q, k, v projections (bias folded at writeback) ----------
  gemm_step<1>(sB, wq, wk, sA, sWt, bq, nullptr, len);   // q -> sA
  gemm_step<1>(sB, wk, wv, sC, sWt, bk, nullptr, len);   // k -> sC
  gemm_step<2>(sB, wv, wo, sVt, sWt, bv, nullptr, len);  // v -> sVt (transposed)
  __syncthreads();

  // ---------- attention: wave -> head = w>>1, M-tiles {(w&1)*2, +1} ----------
  // k/B fragments are shared across the wave's two M-tiles.
  {
    _Float16* aw = sAttn + wv_ * SAW;
    const int head = wv_ >> 1;
    const int MtB  = (wv_ & 1) * 2;

    Acc sc[2][4];
    #pragma unroll
    for (int mi = 0; mi < 2; ++mi)
      #pragma unroll
      for (int i = 0; i < 4; ++i)
        #pragma unroll
        for (int j = 0; j < 8; ++j) sc[mi][i].f[j] = 0.f;

    // scores = q @ k^T for both M-tiles (shared k fragments)
    #pragma unroll
    for (int kc = 0; kc < 2; ++kc) {
      v16h fa0 = load_frag(sA + ((MtB + 0) * 16 + lmod) * SD + head * HDIM + kc * 32 + koff);
      v16h fa1 = load_frag(sA + ((MtB + 1) * 16 + lmod) * SD + head * HDIM + kc * 32 + koff);
      #pragma unroll
      for (int nt = 0; nt < 4; ++nt) {
        v16h fb = load_frag(sC + (nt * 16 + lmod) * SD + head * HDIM + kc * 32 + koff);
        sc[0][nt].v = __builtin_amdgcn_wmma_f32_16x16x32_f16(
            false, fa0, false, fb, (short)0, sc[0][nt].v, false, false);
        sc[1][nt].v = __builtin_amdgcn_wmma_f32_16x16x32_f16(
            false, fa1, false, fb, (short)0, sc[1][nt].v, false, false);
      }
    }

    #pragma unroll
    for (int mi = 0; mi < 2; ++mi) {
      const int Mt = MtB + mi;

      // scale + key-padding mask (column = key index = nt*16 + lmod)
      #pragma unroll
      for (int nt = 0; nt < 4; ++nt) {
        const bool kval = (nt * 16 + lmod) < len;
        #pragma unroll
        for (int r = 0; r < 8; ++r)
          sc[mi][nt].f[r] = kval ? sc[mi][nt].f[r] * 0.125f : NEGBIG;
      }

      // row softmax: row values live across lanes 0-15 within each half-wave
      #pragma unroll
      for (int r = 0; r < 8; ++r) {
        float m = fmaxf(fmaxf(sc[mi][0].f[r], sc[mi][1].f[r]),
                        fmaxf(sc[mi][2].f[r], sc[mi][3].f[r]));
        #pragma unroll
        for (int off = 1; off < 16; off <<= 1) m = fmaxf(m, __shfl_xor(m, off));
        float ssum = 0.f;
        #pragma unroll
        for (int nt = 0; nt < 4; ++nt) {
          const float e = __expf(sc[mi][nt].f[r] - m);
          sc[mi][nt].f[r] = e;
          ssum += e;
        }
        #pragma unroll
        for (int off = 1; off < 16; off <<= 1) ssum += __shfl_xor(ssum, off);
        const float inv = 1.f / ssum;
        #pragma unroll
        for (int nt = 0; nt < 4; ++nt) sc[mi][nt].f[r] *= inv;
      }

      // re-layout attn (C layout -> A layout) via per-wave LDS scratch
      #pragma unroll
      for (int nt = 0; nt < 4; ++nt)
        #pragma unroll
        for (int r = 0; r < 8; ++r)
          aw[(lhalf * 8 + r) * SVD + nt * 16 + lmod] = (_Float16)sc[mi][nt].f[r];

      // ctx = attn @ v
      Acc cx[4];
      #pragma unroll
      for (int i = 0; i < 4; ++i)
        #pragma unroll
        for (int j = 0; j < 8; ++j) cx[i].f[j] = 0.f;

      #pragma unroll
      for (int kc = 0; kc < 2; ++kc) {
        v16h fa = load_frag(aw + lmod * SVD + kc * 32 + koff);
        #pragma unroll
        for (int nt = 0; nt < 4; ++nt) {
          v16h fb = load_frag(sVt + (head * HDIM + nt * 16 + lmod) * SVD + kc * 32 + koff);
          cx[nt].v = __builtin_amdgcn_wmma_f32_16x16x32_f16(
              false, fa, false, fb, (short)0, cx[nt].v, false, false);
        }
      }

      // ctx -> sB (h is dead)
      #pragma unroll
      for (int nt = 0; nt < 4; ++nt)
        #pragma unroll
        for (int r = 0; r < 8; ++r)
          sB[(Mt * 16 + lhalf * 8 + r) * SD + head * HDIM + nt * 16 + lmod] =
              (_Float16)cx[nt].f[r];
    }
  }
  __syncthreads();

  // ---------- out = ctx @ wo + bo, fused masked mean/max pooling ----------
  gemm_step<3>(sB, wo, g1w, nullptr, sWt, bo, sPool, len);
  __syncthreads();

  // ---------- global MLP: g=[mean|max]; g1=relu(g@g1w+g1b); g2=g1@g2w+g2b ----------
  const float invLen = 1.f / (float)len;
  float a1 = g1b[c];
  for (int j = 0; j < DD; ++j)
    a1 = fmaf(sPool[j] * invLen, g1w[j * DD + c], a1);
  for (int j = 0; j < DD; ++j)
    a1 = fmaf(sPool[DD + j], g1w[(DD + j) * DD + c], a1);
  a1 = fmaxf(a1, 0.f);
  sG[c] = a1;          // sU reuse: weight staging done (barrier above)
  __syncthreads();

  float a2 = g2b[c];
  for (int j = 0; j < DD; ++j)
    a2 = fmaf(sG[j], g2w[j * DD + c], a2);

  // ---------- LayerNorm over 256 (one value per thread) ----------
  float s = a2;
  #pragma unroll
  for (int off = 16; off >= 1; off >>= 1) s += __shfl_xor(s, off);
  if (lane == 0) sRed[wv_] = s;
  __syncthreads();
  float tot = 0.f;
  #pragma unroll
  for (int i = 0; i < 8; ++i) tot += sRed[i];
  const float mu = tot * (1.f / 256.f);
  const float d  = a2 - mu;
  float s2 = d * d;
  #pragma unroll
  for (int off = 16; off >= 1; off >>= 1) s2 += __shfl_xor(s2, off);
  if (lane == 0) sRed[8 + wv_] = s2;
  __syncthreads();
  float v2 = 0.f;
  #pragma unroll
  for (int i = 0; i < 8; ++i) v2 += sRed[8 + i];
  const float var = v2 * (1.f / 256.f);

  out[(size_t)b * DD + c] = d * rsqrtf(var + 1e-5f) * ln_g[c] + ln_b[c];
}

extern "C" void kernel_launch(void* const* d_in, const int* in_sizes, int n_in,
                              void* d_out, int out_size, void* d_ws, size_t ws_size,
                              hipStream_t stream) {
  const float* obstacles = (const float*)d_in[0];
  const int*   lengths   = (const int*)d_in[1];
  const float* w1  = (const float*)d_in[2];
  const float* b1  = (const float*)d_in[3];
  const float* w2  = (const float*)d_in[4];
  const float* b2  = (const float*)d_in[5];
  const float* pos = (const float*)d_in[6];
  const float* wq  = (const float*)d_in[7];
  const float* bq  = (const float*)d_in[8];
  const float* wk  = (const float*)d_in[9];
  const float* bk  = (const float*)d_in[10];
  const float* wv  = (const float*)d_in[11];
  const float* bv  = (const float*)d_in[12];
  const float* wo  = (const float*)d_in[13];
  const float* bo  = (const float*)d_in[14];
  const float* g1w = (const float*)d_in[15];
  const float* g1b = (const float*)d_in[16];
  const float* g2w = (const float*)d_in[17];
  const float* g2b = (const float*)d_in[18];
  const float* ln_g = (const float*)d_in[19];
  const float* ln_b = (const float*)d_in[20];
  float* out = (float*)d_out;

  const int B = in_sizes[1];   // lengths has B elements
  obstacle_encoder_kernel<<<dim3(B), dim3(256), 0, stream>>>(
      obstacles, lengths, w1, b1, w2, b2, pos,
      wq, bq, wk, bk, wv, bv, wo, bo,
      g1w, g1b, g2w, g2b, ln_g, ln_b, out);
}
